// AttentionModule_16733192585265
// MI455X (gfx1250) — matrix operand order repaired and checked by reading, not compiled, not run on hardware
//
#include <hip/hip_runtime.h>

typedef __attribute__((ext_vector_type(16))) _Float16 v16h;
typedef __attribute__((ext_vector_type(8)))  _Float16 v8h;
typedef __attribute__((ext_vector_type(4)))  _Float16 v4h;
typedef __attribute__((ext_vector_type(8)))  float    v8f;
typedef __attribute__((ext_vector_type(4)))  float    f32x4;

#define HEADS 128
#define DDIM  128
#define EDIM  768
#define BSZ   16
#define LLEN  64
#define EPAD  8                 // 16B row pad -> LDS row stride 388 dwords (conflict-free)
#define NEG_SLOPE 0.4f
#define MASK_FILL -1e8f

// ---- fast tanh: gfx1250 V_TANH_F32 (confirmed lowering), fallback branch-free ----
__device__ __forceinline__ float fast_tanh(float x) {
#if __has_builtin(__builtin_amdgcn_tanhf)
    return __builtin_amdgcn_tanhf(x);
#else
    float e = __expf(2.0f * x);
    return 1.0f - 2.0f / (e + 1.0f);
#endif
}

// ---- A-matrix 16x32 f16 operand (contiguous k order, wave32): lane t holds row M = t%16.
//   lanes 0-15 : K 0..15 ; lanes 16-31: K 16..31
__device__ __forceinline__ v16h loadA_f16(const float* __restrict__ row, int e0, bool hi) {
    const int off = e0 + (hi ? 16 : 0);
    f32x4 v0 = *(const f32x4*)(row + off);
    f32x4 v1 = *(const f32x4*)(row + off + 4);
    f32x4 v2 = *(const f32x4*)(row + off + 8);
    f32x4 v3 = *(const f32x4*)(row + off + 12);
    v16h r;
#pragma unroll
    for (int i = 0; i < 4; ++i) {
        r[i]      = (_Float16)v0[i];
        r[4 + i]  = (_Float16)v1[i];
        r[8 + i]  = (_Float16)v2[i];
        r[12 + i] = (_Float16)v3[i];
    }
    return r;
}

// ---- B-matrix 32x16 f16 operand from LDS: lane t holds column N = t%16 (= bank row l),
//   lanes 0-15 hold K 0..15, lanes 16-31 hold K 16..31 -> 16 contiguous f16 (2x ds_load_b128)
__device__ __forceinline__ v16h loadB_lds(const _Float16* __restrict__ row, int e0, bool hi) {
    const _Float16* p = row + e0 + (hi ? 16 : 0);
    v8h lo  = *(const v8h*)(p);
    v8h hi8 = *(const v8h*)(p + 8);
    v16h r;
#pragma unroll
    for (int i = 0; i < 8; ++i) { r[i] = lo[i]; r[8 + i] = hi8[i]; }
    return r;
}

// ---- B-matrix 32x16 f16 operand from global (used by qproj for x) ----
__device__ __forceinline__ v16h loadB_f16(const float* __restrict__ row, int e0, bool hi) {
    const float* p = row + e0 + (hi ? 16 : 0);
    f32x4 v0 = *(const f32x4*)(p);
    f32x4 v1 = *(const f32x4*)(p + 4);
    f32x4 v2 = *(const f32x4*)(p + 8);
    f32x4 v3 = *(const f32x4*)(p + 12);
    v16h r;
#pragma unroll
    for (int i = 0; i < 4; ++i) {
        r[i]      = (_Float16)v0[i];
        r[4 + i]  = (_Float16)v1[i];
        r[8 + i]  = (_Float16)v2[i];
        r[12 + i] = (_Float16)v3[i];
    }
    return r;
}

// ---------------- Kernel 1: q = tanh(Query @ x^T), stored as (H, D, B) ----------------
__global__ __launch_bounds__(256)
void qproj_kernel(const float* __restrict__ x,      // (B, E)
                  const float* __restrict__ Query,  // (H, D, E)
                  float* __restrict__ qws) {        // (H, D, B)
    const int h     = blockIdx.x;
    const int t     = threadIdx.x & 31;
    const int wave  = threadIdx.x >> 5;
    const int dBase = wave * 16;
    const int m     = t & 15;
    const bool hi   = (t >= 16);

    const float* aRow = Query + ((size_t)h * DDIM + dBase + m) * EDIM;  // row M = m
    const float* bRow = x + (size_t)m * EDIM;                           // column N = b = m

    v8f c = {};
    for (int e0 = 0; e0 < EDIM; e0 += 32) {
        v16h A = loadA_f16(aRow, e0, hi);
        v16h B = loadB_f16(bRow, e0, hi);
        c = __builtin_amdgcn_wmma_f32_16x16x32_f16(false, A, false, B,
                                                   (short)0, c, false, false);
    }

    float* qout = qws + ((size_t)h * DDIM + dBase + (hi ? 8 : 0)) * BSZ + m;
#pragma unroll
    for (int r = 0; r < 8; ++r)
        qout[(size_t)r * BSZ] = fast_tanh(c[r]);
}

// --------- Kernel 2: fused  k=tanh(Key@bank^T), score, softmax, emb, leaky-relu ---------
__global__ __launch_bounds__(256)
void attn_kernel(const float* __restrict__ bank,  // (B, L, E)
                 const int*   __restrict__ mask,  // (B, L)
                 const float* __restrict__ Key,   // (H, D, E)
                 const float* __restrict__ qws,   // (H, D, B)
                 float* __restrict__ out) {       // (B, H, E)
    __shared__ __align__(16) _Float16 bankLds[LLEN][EDIM + EPAD];   // ~97 KB (of 320 KB WGP LDS)
    __shared__ float scoreLds[LLEN];
    __shared__ float attnLds[LLEN];

    const int h   = blockIdx.x >> 4;
    const int b   = blockIdx.x & 15;
    const int tid = threadIdx.x;
    const int t   = tid & 31;
    const int wave  = tid >> 5;
    const int dBase = wave * 16;
    const int m     = t & 15;
    const bool hi   = (t >= 16);

    const float* bankB = bank + (size_t)b * LLEN * EDIM;

    // ---- stage bank[b] into LDS as f16 (once per workgroup), conflict-free 8B stores ----
    for (int l = wave; l < LLEN; l += 8) {
        const float* src = bankB + (size_t)l * EDIM;
#pragma unroll
        for (int e = t * 4; e < EDIM; e += 32 * 4) {
            f32x4 v = *(const f32x4*)(src + e);
            v4h hv;
#pragma unroll
            for (int i = 0; i < 4; ++i) hv[i] = (_Float16)v[i];
            *(v4h*)&bankLds[l][e] = hv;
        }
    }
    if (tid < LLEN) scoreLds[tid] = 0.0f;
    __syncthreads();

    // ---- WMMA GEMM: wave owns D rows [dBase, dBase+16) x all 64 L columns -> 4 tiles ----
    const float* aRow = Key + ((size_t)h * DDIM + dBase + m) * EDIM;   // A row M = m
    const _Float16* lRow0 = &bankLds[0 * 16 + m][0];
    const _Float16* lRow1 = &bankLds[1 * 16 + m][0];
    const _Float16* lRow2 = &bankLds[2 * 16 + m][0];
    const _Float16* lRow3 = &bankLds[3 * 16 + m][0];

    v8f c0 = {}, c1 = {}, c2 = {}, c3 = {};
    for (int e0 = 0; e0 < EDIM; e0 += 32) {
        v16h A  = loadA_f16(aRow, e0, hi);
        v16h B0 = loadB_lds(lRow0, e0, hi);
        v16h B1 = loadB_lds(lRow1, e0, hi);
        v16h B2 = loadB_lds(lRow2, e0, hi);
        v16h B3 = loadB_lds(lRow3, e0, hi);
        c0 = __builtin_amdgcn_wmma_f32_16x16x32_f16(false, A, false, B0, (short)0, c0, false, false);
        c1 = __builtin_amdgcn_wmma_f32_16x16x32_f16(false, A, false, B1, (short)0, c1, false, false);
        c2 = __builtin_amdgcn_wmma_f32_16x16x32_f16(false, A, false, B2, (short)0, c2, false, false);
        c3 = __builtin_amdgcn_wmma_f32_16x16x32_f16(false, A, false, B3, (short)0, c3, false, false);
    }

    // q values for this lane's 8 D-rows (d = dBase + 8*hi + r), fixed b
    const float* qcol = qws + ((size_t)h * DDIM + dBase + (hi ? 8 : 0)) * BSZ + b;
    float qv[8];
#pragma unroll
    for (int r = 0; r < 8; ++r) qv[r] = qcol[(size_t)r * BSZ];

    // partial score per owned L column: sum_r tanh(k) * q
    float p0 = 0.f, p1 = 0.f, p2 = 0.f, p3 = 0.f;
#pragma unroll
    for (int r = 0; r < 8; ++r) {
        p0 = fmaf(fast_tanh(c0[r]), qv[r], p0);
        p1 = fmaf(fast_tanh(c1[r]), qv[r], p1);
        p2 = fmaf(fast_tanh(c2[r]), qv[r], p2);
        p3 = fmaf(fast_tanh(c3[r]), qv[r], p3);
    }
    atomicAdd(&scoreLds[0 * 16 + m], p0);   // ds_add_f32
    atomicAdd(&scoreLds[1 * 16 + m], p1);
    atomicAdd(&scoreLds[2 * 16 + m], p2);
    atomicAdd(&scoreLds[3 * 16 + m], p3);
    __syncthreads();

    // masked scores
    if (tid < LLEN) {
        float s = (mask[(size_t)b * LLEN + tid] == 0) ? MASK_FILL : scoreLds[tid];
        attnLds[tid] = s;
    }
    __syncthreads();

    // softmax over L=64 (redundantly per thread; trivial cost)
    float mx = -3.4e38f;
#pragma unroll 8
    for (int l = 0; l < LLEN; ++l) mx = fmaxf(mx, attnLds[l]);
    float sum = 0.f;
#pragma unroll 8
    for (int l = 0; l < LLEN; ++l) sum += __expf(attnLds[l] - mx);
    const float inv = 1.0f / sum;
    __syncthreads();
    if (tid < LLEN) attnLds[tid] = __expf(attnLds[tid] - mx) * inv;
    __syncthreads();

    // emb[e] = sum_l attn[l] * bank[b,l,e]; bank served from LDS (f16), f32 accumulate
    float a0 = 0.f, a1 = 0.f, a2 = 0.f;
    for (int l = 0; l < LLEN; ++l) {
        const float w = attnLds[l];
        a0 = fmaf(w, (float)bankLds[l][tid],       a0);
        a1 = fmaf(w, (float)bankLds[l][tid + 256], a1);
        a2 = fmaf(w, (float)bankLds[l][tid + 512], a2);
    }
    float* orow = out + ((size_t)b * HEADS + h) * EDIM + tid;
    float v;
    v = a0; orow[0]   = (v >= 0.f) ? v : NEG_SLOPE * v;
    v = a1; orow[256] = (v >= 0.f) ? v : NEG_SLOPE * v;
    v = a2; orow[512] = (v >= 0.f) ? v : NEG_SLOPE * v;
}

extern "C" void kernel_launch(void* const* d_in, const int* in_sizes, int n_in,
                              void* d_out, int out_size, void* d_ws, size_t ws_size,
                              hipStream_t stream) {
    (void)in_sizes; (void)n_in; (void)out_size; (void)ws_size;
    const float* x     = (const float*)d_in[0];  // (B, E)
    const float* bank  = (const float*)d_in[1];  // (B, L, E)
    const int*   mask  = (const int*)  d_in[2];  // (B, L)
    const float* Query = (const float*)d_in[3];  // (H, D, E)
    const float* Key   = (const float*)d_in[4];  // (H, D, E)
    float* out = (float*)d_out;                  // (B, H, E)
    float* qws = (float*)d_ws;                   // (H, D, B) = 1 MB

    qproj_kernel<<<HEADS, 256, 0, stream>>>(x, Query, qws);
    attn_kernel<<<HEADS * BSZ, 256, 0, stream>>>(bank, mask, Key, qws, out);
}
